// PatchToLatentCrossAttention_81011673137182
// MI455X (gfx1250) — matrix operand order repaired and checked by reading, not compiled, not run on hardware
//
#include <hip/hip_runtime.h>
#include <hip/hip_bf16.h>

typedef __bf16 bf16;
typedef __attribute__((ext_vector_type(16))) __bf16 v16bf;
typedef __attribute__((ext_vector_type(8)))  float  v8f;

#define Bc   4
#define LQc  4096
#define LKVc 256
#define Dc   1024
#define Hc   16
#define KVHc 4
#define HDc  64

// ln(10000)/32
#define ROPE_LOGF_OVER_HALFDIM 0.28782313662425575f

union AFrag { v16bf v; uint4 u[2]; };

__device__ __forceinline__ v8f wmma_bf16f32(v16bf a, v16bf b, v8f c) {
  // D = A(16x32 bf16) * B(32x16 bf16) + C(16x16 f32)
  return __builtin_amdgcn_wmma_f32_16x16x32_bf16(false, a, false, b, (short)0, c, false, false);
}

__device__ __forceinline__ v8f zero8() {
  v8f z = {0.f,0.f,0.f,0.f,0.f,0.f,0.f,0.f};
  return z;
}

// ---------------- elementwise f32 -> bf16 ----------------
__global__ void cvt_f32_bf16_kernel(const float* __restrict__ in, bf16* __restrict__ out, int n) {
  int i = (blockIdx.x * blockDim.x + threadIdx.x) * 4;
  if (i + 3 < n) {
    float4 f = *(const float4*)(in + i);
    union { bf16 h[4]; uint2 u; } t;
    t.h[0] = (bf16)f.x; t.h[1] = (bf16)f.y; t.h[2] = (bf16)f.z; t.h[3] = (bf16)f.w;
    *(uint2*)(out + i) = t.u;
  } else {
    for (; i < n; ++i) out[i] = (bf16)in[i];
  }
}

// ---------------- W (K x N) f32 -> WT (N x K) bf16 ----------------
__global__ void transpose_cvt_kernel(const float* __restrict__ W, bf16* __restrict__ WT,
                                     int Kd, int Nd) {
  int idx = blockIdx.x * blockDim.x + threadIdx.x;
  if (idx < Kd * Nd) {
    int kk = idx / Nd;
    int n  = idx % Nd;
    WT[(size_t)n * Kd + kk] = (bf16)W[idx];
  }
}

// ---------------- GEMM: out(MxN) = A(MxK) * BT(NxK)^T + bias, optional RoPE ----------------
// Block = 128 threads = 4 waves. Block tile 64(M) x 64(N). Each wave: 16(M) x 64(N).
// K-loop is double-buffered (K is a multiple of 64) to hide vmem latency.
template<bool ROPE, bool OUT_F32>
__global__ void __launch_bounds__(128, 1)
gemm_bf16_kernel(const bf16* __restrict__ A, const bf16* __restrict__ BT,
                 const float* __restrict__ bias, void* __restrict__ outp,
                 int M, int N, int K, int L, const int* __restrict__ nf_ptr) {
  const int wave   = threadIdx.x >> 5;
  const int lane   = threadIdx.x & 31;
  const int lanelo = lane & 15;
  const bool hi    = lane >= 16;
  const int m0 = blockIdx.x * 64 + wave * 16;
  const int n0 = blockIdx.y * 64;

  v8f acc[4];
#pragma unroll
  for (int t = 0; t < 4; ++t) acc[t] = zero8();

  const bf16* arow  = A + (size_t)(m0 + lanelo) * K + (hi ? 16 : 0);
  const bf16* brow0 = BT + (size_t)(n0 + lanelo) * K + (hi ? 16 : 0);

  auto loadA = [&](AFrag& a, int k0) {
    a.u[0] = *(const uint4*)(arow + k0);        // k0(+16 hi)   .. +8
    a.u[1] = *(const uint4*)(arow + k0 + 8);    // k0+8(+16hi) .. +8
  };
  auto loadB = [&](v16bf* bb, int k0) {
#pragma unroll
    for (int t = 0; t < 4; ++t)
      bb[t] = *(const v16bf*)(brow0 + (size_t)t * 16 * K + k0);  // 16 contiguous K values
  };

  AFrag a0, a1;
  v16bf b0[4], b1[4];
  loadA(a0, 0);
  loadB(b0, 0);
  for (int k0 = 0; k0 < K; k0 += 64) {
    loadA(a1, k0 + 32);
    loadB(b1, k0 + 32);
    if (k0 + 512 < K) __builtin_prefetch(arow + k0 + 512, 0, 3);  // global_prefetch_b8
#pragma unroll
    for (int t = 0; t < 4; ++t) acc[t] = wmma_bf16f32(a0.v, b0[t], acc[t]);
    if (k0 + 64 < K) {
      loadA(a0, k0 + 64);
      loadB(b0, k0 + 64);
    }
#pragma unroll
    for (int t = 0; t < 4; ++t) acc[t] = wmma_bf16f32(a1.v, b1[t], acc[t]);
  }

  // bias (reference adds bias before RoPE)
#pragma unroll
  for (int t = 0; t < 4; ++t) {
    float bv = bias[n0 + t * 16 + lanelo];
#pragma unroll
    for (int r = 0; r < 8; ++r) acc[t][r] += bv;
  }

  if (ROPE) {
    int nf = *nf_ptr;
    if (nf > 1) {
      int tpf = L / nf;  // tokens per frame
#pragma unroll
      for (int tp = 0; tp < 2; ++tp) {
        // dn = tp*16+lanelo (first half) pairs with dn+32 (tile tp+2); shared freq index j
        float jv   = (float)(tp * 16 + lanelo);
        float invf = __expf(-jv * ROPE_LOGF_OVER_HALFDIM);
#pragma unroll
        for (int r = 0; r < 8; ++r) {
          int m   = m0 + r + (hi ? 8 : 0);
          int tok = m % L;
          int pos = tok / tpf;
          float ang = (float)pos * invf;
          float c = __cosf(ang), s = __sinf(ang);
          float v0 = acc[tp][r];
          float v1 = acc[tp + 2][r];
          acc[tp][r]     = v0 * c - v1 * s;   // x1*cos - x2*sin
          acc[tp + 2][r] = v1 * c + v0 * s;   // x2*cos + x1*sin
        }
      }
    }
  }

#pragma unroll
  for (int t = 0; t < 4; ++t) {
    int n = n0 + t * 16 + lanelo;
#pragma unroll
    for (int r = 0; r < 8; ++r) {
      int m = m0 + r + (hi ? 8 : 0);
      if (OUT_F32) ((float*)outp)[(size_t)m * N + n] = acc[t][r];
      else         ((bf16*)outp)[(size_t)m * N + n]  = (bf16)acc[t][r];
    }
  }
}

// ---------------- attention: per (b, h, 64-query chunk) ----------------
// 128 threads = 4 waves; each wave handles 16 query rows.
// K tile is staged in LDS via gfx1250 async global->LDS copies (ASYNCcnt);
// V is staged transposed (element transpose, plain stores); flash-style
// online softmax over 4 KV chunks of 64 keeps register pressure low.
#define KL_STRIDE 64    // K LDS row stride (elems): rows are 128B, 32B aligned
#define VT_STRIDE 272   // padded KV stride for transposed V in LDS (row bytes % 32 == 0)
#define P_STRIDE  72    // padded chunk stride for probs in LDS (row bytes % 16 == 0)
#define KL_BYTES  (LKVc * KL_STRIDE * 2)            // 32768
#define VT_BYTES  (HDc * VT_STRIDE * 2)             // 34816
#define ATTN_LDS_BYTES (KL_BYTES + VT_BYTES + 4 * 16 * P_STRIDE * 2)

__global__ void __launch_bounds__(128, 1)
attn_kernel(const bf16* __restrict__ q, const bf16* __restrict__ k,
            const bf16* __restrict__ v, bf16* __restrict__ attn) {
  extern __shared__ char smem[];
  bf16* Klds  = (bf16*)smem;                        // [LKVc][KL_STRIDE]
  bf16* Vt    = (bf16*)(smem + KL_BYTES);           // [HDc][VT_STRIDE]  V transposed: Vt[d][kv]
  bf16* probs = (bf16*)(smem + KL_BYTES + VT_BYTES);// [4 waves][16][P_STRIDE]

  const int b  = blockIdx.z;
  const int h  = blockIdx.y;
  const int hv = h / (Hc / KVHc);

  // ---- async stage K (strided 2D region -> dense LDS rows), 16B per lane ----
  {
    const bf16* kgbase = k + (size_t)b * LKVc * (KVHc * HDc) + hv * HDc;
    unsigned klds0 = (unsigned)(size_t)Klds;   // low 32 bits of generic addr == LDS offset
    for (int idx = threadIdx.x; idx < LKVc * 4; idx += blockDim.x) {
      int kv = idx >> 2;
      int ch = idx & 3;                        // 4 x 16B chunks per 128B row
      const bf16* g = kgbase + (size_t)kv * (KVHc * HDc) + ch * 8;
      unsigned l = klds0 + (unsigned)((kv * KL_STRIDE + ch * 8) * 2);
      asm volatile("global_load_async_to_lds_b128 %0, %1, off"
                   :: "v"(l), "v"(g) : "memory");
    }
  }

  // ---- stage V transposed into LDS (element-granularity transpose) ----
  const bf16* vg = v + (size_t)b * LKVc * (KVHc * HDc) + hv * HDc;
  for (int idx = threadIdx.x; idx < LKVc * HDc; idx += blockDim.x) {
    int kv = idx >> 6;
    int d  = idx & 63;
    Vt[d * VT_STRIDE + kv] = vg[(size_t)kv * (KVHc * HDc) + d];
  }
  asm volatile("s_wait_asynccnt 0x0" ::: "memory");  // my async K chunks are in LDS
  __syncthreads();                                   // everyone's chunks visible

  const int wave   = threadIdx.x >> 5;
  const int lane   = threadIdx.x & 31;
  const int lanelo = lane & 15;
  const bool hi    = lane >= 16;
  const int m0     = blockIdx.x * 64 + wave * 16;   // query row base within LQ
  bf16* pw = probs + wave * 16 * P_STRIDE;

  // Q fragments for this wave's 16 rows (K-dim = HD = 64 -> 2 WMMA k-steps)
  AFrag aq[2];
  const bf16* qrow = q + ((size_t)b * LQc + m0 + lanelo) * Dc + h * HDc + (hi ? 16 : 0);
#pragma unroll
  for (int ks = 0; ks < 2; ++ks) {
    aq[ks].u[0] = *(const uint4*)(qrow + ks * 32);
    aq[ks].u[1] = *(const uint4*)(qrow + ks * 32 + 8);
  }

  v8f o[4];
#pragma unroll
  for (int nt = 0; nt < 4; ++nt) o[nt] = zero8();
  float Mr[8], Sr[8];
#pragma unroll
  for (int r = 0; r < 8; ++r) { Mr[r] = -3.0e38f; Sr[r] = 0.f; }

  const float scale = 0.125f;   // 1/sqrt(HD)

  for (int c = 0; c < 4; ++c) {          // KV chunks of 64
    // ---- scores for this chunk: 4 tiles of 16x16 (K fragments from LDS) ----
    v8f s4[4];
#pragma unroll
    for (int t = 0; t < 4; ++t) s4[t] = zero8();
#pragma unroll
    for (int ks = 0; ks < 2; ++ks) {
      v16bf bk[4];
#pragma unroll
      for (int t = 0; t < 4; ++t) {
        const bf16* krow = Klds + (c * 64 + t * 16 + lanelo) * KL_STRIDE
                                + ks * 32 + (hi ? 16 : 0);
        bk[t] = *(const v16bf*)krow;
      }
#pragma unroll
      for (int t = 0; t < 4; ++t) s4[t] = wmma_bf16f32(aq[ks].v, bk[t], s4[t]);
    }

    // ---- online softmax update (rows: lanes 0-15 -> m=r, lanes 16-31 -> m=r+8) ----
#pragma unroll
    for (int r = 0; r < 8; ++r) {
      float cm = -3.0e38f;
#pragma unroll
      for (int t = 0; t < 4; ++t) cm = fmaxf(cm, s4[t][r]);
      for (int off = 8; off > 0; off >>= 1) cm = fmaxf(cm, __shfl_xor(cm, off, 32));
      float Mn = fmaxf(Mr[r], cm);
      float f  = __expf((Mr[r] - Mn) * scale);
      float ps = 0.f;
#pragma unroll
      for (int t = 0; t < 4; ++t) {
        float e = __expf((s4[t][r] - Mn) * scale);
        s4[t][r] = e;
        ps += e;
      }
      for (int off = 8; off > 0; off >>= 1) ps += __shfl_xor(ps, off, 32);
      Mr[r] = Mn;
      Sr[r] = Sr[r] * f + ps;
#pragma unroll
      for (int nt = 0; nt < 4; ++nt) o[nt][r] *= f;
      int m = r + (hi ? 8 : 0);
#pragma unroll
      for (int t = 0; t < 4; ++t)
        pw[m * P_STRIDE + t * 16 + lanelo] = (bf16)s4[t][r];
    }
    // per-wave LDS buffer: wave-internal ds ordering covers the write->read RAW

    // ---- o += P(16x64) * V_chunk(64x64) ----
#pragma unroll
    for (int ks = 0; ks < 2; ++ks) {
      AFrag ap;
      const bf16* prow = pw + lanelo * P_STRIDE + ks * 32 + (hi ? 16 : 0);
      ap.u[0] = *(const uint4*)(prow);
      ap.u[1] = *(const uint4*)(prow + 8);
      v16bf bv[4];
#pragma unroll
      for (int nt = 0; nt < 4; ++nt) {
        const bf16* vrow = Vt + (nt * 16 + lanelo) * VT_STRIDE
                              + c * 64 + ks * 32 + (hi ? 16 : 0);
        bv[nt] = *(const v16bf*)vrow;
      }
#pragma unroll
      for (int nt = 0; nt < 4; ++nt) o[nt] = wmma_bf16f32(ap.v, bv[nt], o[nt]);
    }
  }

  // ---- normalize and store attn_out (B, LQ, H*HD) bf16 ----
  bf16* og = attn + ((size_t)b * LQc + m0) * Dc + h * HDc;
#pragma unroll
  for (int r = 0; r < 8; ++r) {
    float inv = 1.0f / Sr[r];
#pragma unroll
    for (int nt = 0; nt < 4; ++nt) {
      og[(size_t)(r + (hi ? 8 : 0)) * Dc + nt * 16 + lanelo] = (bf16)(o[nt][r] * inv);
    }
  }
}

// ---------------- launch ----------------
extern "C" void kernel_launch(void* const* d_in, const int* in_sizes, int n_in,
                              void* d_out, int out_size, void* d_ws, size_t ws_size,
                              hipStream_t stream) {
  (void)in_sizes; (void)n_in; (void)out_size; (void)ws_size;
  const float* patches = (const float*)d_in[0];
  const float* latents = (const float*)d_in[1];
  const float* Wq = (const float*)d_in[2];
  const float* bq = (const float*)d_in[3];
  const float* Wk = (const float*)d_in[4];
  const float* bk = (const float*)d_in[5];
  const float* Wv = (const float*)d_in[6];
  const float* bv = (const float*)d_in[7];
  const float* Wo = (const float*)d_in[8];
  const float* bo = (const float*)d_in[9];
  const int*   nf = (const int*)d_in[10];
  float* out = (float*)d_out;

  const int Mq = Bc * LQc;    // 16384
  const int Mk = Bc * LKVc;   // 1024
  const int Nq = Hc * HDc;    // 1024
  const int Nkv = KVHc * HDc; // 256

  char* ws = (char*)d_ws;
  size_t off = 0;
  auto alloc = [&](size_t bytes) -> void* {
    void* p = ws + off;
    off = (off + bytes + 255) & ~(size_t)255;
    return p;
  };
  bf16* p_bf  = (bf16*)alloc((size_t)Mq * Dc * 2);
  bf16* l_bf  = (bf16*)alloc((size_t)Mk * Dc * 2);
  bf16* WqT   = (bf16*)alloc((size_t)Dc * Nq * 2);
  bf16* WkT   = (bf16*)alloc((size_t)Dc * Nkv * 2);
  bf16* WvT   = (bf16*)alloc((size_t)Dc * Nkv * 2);
  bf16* WoT   = (bf16*)alloc((size_t)Dc * Dc * 2);
  bf16* qbuf  = (bf16*)alloc((size_t)Mq * Nq * 2);
  bf16* kbuf  = (bf16*)alloc((size_t)Mk * Nkv * 2);
  bf16* vbuf  = (bf16*)alloc((size_t)Mk * Nkv * 2);
  bf16* abuf  = (bf16*)alloc((size_t)Mq * Dc * 2);

  // stage 0: convert activations + weights to bf16 (weights transposed to N x K)
  {
    int n0 = Mq * Dc;
    cvt_f32_bf16_kernel<<<(n0 / 4 + 255) / 256, 256, 0, stream>>>(patches, p_bf, n0);
    int n1 = Mk * Dc;
    cvt_f32_bf16_kernel<<<(n1 / 4 + 255) / 256, 256, 0, stream>>>(latents, l_bf, n1);
    transpose_cvt_kernel<<<(Dc * Nq + 255) / 256, 256, 0, stream>>>(Wq, WqT, Dc, Nq);
    transpose_cvt_kernel<<<(Dc * Nkv + 255) / 256, 256, 0, stream>>>(Wk, WkT, Dc, Nkv);
    transpose_cvt_kernel<<<(Dc * Nkv + 255) / 256, 256, 0, stream>>>(Wv, WvT, Dc, Nkv);
    transpose_cvt_kernel<<<(Dc * Dc + 255) / 256, 256, 0, stream>>>(Wo, WoT, Dc, Dc);
  }

  // stage 1: Q/K/V projections (+bias, +RoPE for Q,K)
  gemm_bf16_kernel<true, false><<<dim3(Mq / 64, Nq / 64), 128, 0, stream>>>(
      p_bf, WqT, bq, qbuf, Mq, Nq, Dc, LQc, nf);
  gemm_bf16_kernel<true, false><<<dim3(Mk / 64, Nkv / 64), 128, 0, stream>>>(
      l_bf, WkT, bk, kbuf, Mk, Nkv, Dc, LKVc, nf);
  gemm_bf16_kernel<false, false><<<dim3(Mk / 64, Nkv / 64), 128, 0, stream>>>(
      l_bf, WvT, bv, vbuf, Mk, Nkv, Dc, 1, nf);

  // stage 2: GQA attention (flash-style, async K staging)
  attn_kernel<<<dim3(LQc / 64, Hc, Bc), 128, ATTN_LDS_BYTES, stream>>>(qbuf, kbuf, vbuf, abuf);

  // stage 3: output projection -> f32 d_out
  gemm_bf16_kernel<false, true><<<dim3(Mq / 64, Dc / 64), 128, 0, stream>>>(
      abuf, WoT, bo, out, Mq, Dc, Dc, 1, nf);
}
